// CNN_80221399155117
// MI455X (gfx1250) — matrix operand order repaired and checked by reading, not compiled, not run on hardware
//
#include <hip/hip_runtime.h>
#include <stdint.h>

#define B_      4
#define S_      1024
#define V_      32000
#define H_      2048
#define OC_     128
#define KW_     5
#define POOLED_ 204
#define FLAT_   26115
#define CP_     2112      // H+1 padded up to 33 K-chunks of 64
#define CT2_    33        // K-chunks of 64 per tap
#define NCHUNK_ (KW_ * CT2_)   // 165

typedef __attribute__((ext_vector_type(16))) __bf16 bf16x16;
typedef __attribute__((ext_vector_type(8)))  float  floatx8;
typedef __attribute__((ext_vector_type(4)))  float  floatx4;   // clang vector (nontemporal-ok)

union BFrag { bf16x16 v; uint4 q[2]; };

__device__ __forceinline__ unsigned short f2bf(float f) {
  union { float f; unsigned int u; } c; c.f = f;
  unsigned int u = c.u;
  unsigned int r = (u + 0x7fffu + ((u >> 16) & 1u)) >> 16;
  return (unsigned short)r;
}

__device__ __forceinline__ void online_upd(float x, float& m, float& s) {
  const float L2E = 1.4426950408889634f;
  float t = x * L2E;
  float mn = fmaxf(m, t);
  s = s * exp2f(m - mn) + exp2f(t - mn);
  m = mn;
}

// async copy 16B global -> LDS (ASYNCcnt), gfx1250
__device__ __forceinline__ void async_b128(unsigned lds_off, const void* gptr) {
  unsigned long long ga = (unsigned long long)(uintptr_t)gptr;
  asm volatile("global_load_async_to_lds_b128 %0, %1, off"
               :: "v"(lds_off), "v"(ga) : "memory");
}
__device__ __forceinline__ void wait_async0() {
  asm volatile("s_wait_asynccnt 0" ::: "memory");
}

// ---------------- kernel 1: fused softmax -> gather -> -log2 (surprisal) ----
__global__ __launch_bounds__(256) void k_surprisal(
    const float* __restrict__ logits, const int* __restrict__ ids,
    const float* __restrict__ mask, float* __restrict__ surp) {
  const int row = blockIdx.x;           // b*S + s, 4096 rows
  const int tid = threadIdx.x;
  const floatx4* lp = (const floatx4*)(logits + (size_t)row * V_);
  float m = -3.0e38f, s = 0.f;
  for (int i = tid; i < V_ / 4; i += 256) {
    floatx4 v = __builtin_nontemporal_load(lp + i);  // 512MB single-use stream
    online_upd(v.x, m, s);
    online_upd(v.y, m, s);
    online_upd(v.z, m, s);
    online_upd(v.w, m, s);
  }
  // wave32 reduction of (m, s)
  for (int off = 16; off; off >>= 1) {
    float om = __shfl_down(m, off);
    float os = __shfl_down(s, off);
    float mn = fmaxf(m, om);
    s = s * exp2f(m - mn) + os * exp2f(om - mn);
    m = mn;
  }
  __shared__ float sm[8], ss[8];
  if ((tid & 31) == 0) { sm[tid >> 5] = m; ss[tid >> 5] = s; }
  __syncthreads();
  if (tid == 0) {
    float fm = sm[0], fs = ss[0];
    for (int w = 1; w < 8; ++w) {
      float mn = fmaxf(fm, sm[w]);
      fs = fs * exp2f(fm - mn) + ss[w] * exp2f(sm[w] - mn);
      fm = mn;
    }
    float lt = logits[(size_t)row * V_ + ids[row]];
    // -log2(p) = log2(sum 2^t) - t_tok   (t = logit * log2(e))
    surp[row] = (fm + log2f(fs) - lt * 1.4426950408889634f) * mask[row];
  }
}

// ---------------- kernel 2: feats -> bf16 [B][S][CP_] (B-operand layout) ----
__global__ __launch_bounds__(256) void k_pack_x(
    const float* __restrict__ hidden, const float* __restrict__ surp,
    unsigned short* __restrict__ xb) {
  int idx = blockIdx.x * 256 + threadIdx.x;
  const int total = B_ * S_ * CP_;
  if (idx >= total) return;
  int c = idx % CP_;
  int bs = idx / CP_;
  float v = 0.f;
  if (c < H_)       v = hidden[(size_t)bs * H_ + c];
  else if (c == H_) v = surp[bs];
  xb[idx] = f2bf(v);
}

// ---------------- kernel 3: conv_w -> bf16 [k][OC][CP_] (A-operand) --------
__global__ __launch_bounds__(256) void k_pack_w(
    const float* __restrict__ cw, unsigned short* __restrict__ wA) {
  int idx = blockIdx.x * 256 + threadIdx.x;
  const int total = KW_ * OC_ * CP_;
  if (idx >= total) return;
  int c = idx % CP_;
  int r = idx / CP_;            // r = k*OC_ + oc
  int k = r / OC_;
  int oc = r - k * OC_;
  float v = 0.f;
  if (c < H_ + 1) v = cw[((size_t)oc * (H_ + 1) + c) * KW_ + k];
  wA[idx] = f2bf(v);
}

// ---------------- kernel 4: implicit-im2col conv GEMM via WMMA bf16 --------
// y[b][oc][s] = bias[oc] + sum_k sum_c wA[k][oc][c] * xb[b][s+k-2][c]
// Block: 256 thr = 8 waves; tile M=128 (wave w -> oc 16w..16w+15), N=32.
// K-chunk = 64 channels; double-buffered LDS fed by async global->LDS copies.
__global__ __launch_bounds__(256) void k_conv_wmma(
    const unsigned short* __restrict__ xb, const unsigned short* __restrict__ wA,
    const float* __restrict__ bias, float* __restrict__ y) {
  __shared__ unsigned short lw[2][OC_ * 64];   // W tile: 128 oc x 64 c (16KB)
  __shared__ unsigned short lx[2][32 * 64];    // X tile: 32 cols x 64 c (4KB)

  const int tid  = threadIdx.x;
  const int wave = tid >> 5;
  const int lane = tid & 31;
  const int b    = blockIdx.y;
  const int s0   = blockIdx.x * 32;

  const int wOc = tid >> 1, wHalf = tid & 1;   // W staging: 64B / thread
  const int xRow = tid >> 3, xSeg = tid & 7;   // X staging: 16B / thread

  floatx8 acc0 = {0.f,0.f,0.f,0.f,0.f,0.f,0.f,0.f};
  floatx8 acc1 = {0.f,0.f,0.f,0.f,0.f,0.f,0.f,0.f};

  auto fetch_async = [&](int chunk, int buf) {
    int k = chunk / CT2_, ct = chunk - k * CT2_;   // c0 = ct*64
    // W tile: 4x async b128 per thread
    const unsigned short* wsrc =
        wA + ((size_t)(k * OC_ + wOc) * CP_ + ct * 64 + wHalf * 32);
    unsigned wdst = (unsigned)(uintptr_t)&lw[buf][wOc * 64 + wHalf * 32];
    #pragma unroll
    for (int q = 0; q < 4; ++q)
      async_b128(wdst + q * 16, wsrc + q * 8);
    // X tile: 1x async b128 per thread (zero-fill halo rows with ds store)
    int sr = s0 + k - 2 + xRow;                    // pad=2 halo
    unsigned short* xdst = &lx[buf][xRow * 64 + xSeg * 8];
    if (sr >= 0 && sr < S_) {
      async_b128((unsigned)(uintptr_t)xdst,
                 xb + ((size_t)(b * S_ + sr) * CP_ + ct * 64 + xSeg * 8));
    } else {
      *(uint4*)xdst = make_uint4(0u, 0u, 0u, 0u);
    }
  };

  fetch_async(0, 0);

  // A fragment addressing per ISA 16-bit A 16x32 layout:
  //  lanes 0-15 : elems0-7 = K 0..7 , elems8-15 = K 16..23
  //  lanes16-31 : elems0-7 = K 8..15, elems8-15 = K 24..31
  const int mrow = (wave << 4) + (lane & 15);
  const int qa = (lane < 16) ? 0 : 8;
  // B fragment: lanes 0-15 col n, K 0..7,16..23 ; lanes 16-31 col n, K 8..15,24..31
  const int nb = lane & 15;
  const int qb = (lane < 16) ? 0 : 8;

  int buf = 0;
  for (int chunk = 0; chunk < NCHUNK_; ++chunk) {
    wait_async0();                         // own async group for `buf` done
    __syncthreads();                       // all waves' tiles visible in LDS
    if (chunk + 1 < NCHUNK_) fetch_async(chunk + 1, buf ^ 1);  // overlap DMA

    #pragma unroll
    for (int sub = 0; sub < 2; ++sub) {    // two K=32 steps per chunk
      BFrag A, B0, B1;
      const unsigned short* aw = &lw[buf][mrow * 64 + sub * 32 + qa];
      A.q[0] = *(const uint4*)aw;
      A.q[1] = *(const uint4*)(aw + 16);
      const uint4* b0p = (const uint4*)&lx[buf][nb * 64 + sub * 32 + qb];
      B0.q[0] = b0p[0]; B0.q[1] = b0p[2];
      const uint4* b1p = (const uint4*)&lx[buf][(16 + nb) * 64 + sub * 32 + qb];
      B1.q[0] = b1p[0]; B1.q[1] = b1p[2];

      acc0 = __builtin_amdgcn_wmma_f32_16x16x32_bf16(false, A.v, false, B0.v,
                                                     (short)0, acc0, false, false);
      acc1 = __builtin_amdgcn_wmma_f32_16x16x32_bf16(false, A.v, false, B1.v,
                                                     (short)0, acc1, false, false);
    }
    buf ^= 1;
  }

  // C/D layout: VGPR r -> M = r (lanes 0-15, N=lane) / M = r+8 (lanes 16-31)
  #pragma unroll
  for (int r = 0; r < 8; ++r) {
    int oc = (wave << 4) + r + ((lane >= 16) ? 8 : 0);
    float bv = bias[oc];
    int sc = s0 + (lane & 15);
    float* yp = y + ((size_t)(b * OC_ + oc) << 10);
    yp[sc]      = acc0[r] + bv;
    yp[sc + 16] = acc1[r] + bv;
  }
}

// ---------------- kernel 5: maxpool5 + concat sentiment + relu -> z --------
__global__ __launch_bounds__(256) void k_pool_relu(
    const float* __restrict__ y, const float* __restrict__ sent,
    float* __restrict__ z) {
  int idx = blockIdx.x * 256 + threadIdx.x;
  const int total = B_ * FLAT_;
  if (idx >= total) return;
  int b = idx / FLAT_;
  int j = idx - b * FLAT_;
  float v;
  if (j < OC_ * POOLED_) {
    int oc = j / POOLED_;
    int p  = j - oc * POOLED_;
    const float* yp = y + (((size_t)(b * OC_ + oc)) << 10) + p * 5;
    v = yp[0];
    v = fmaxf(v, yp[1]);
    v = fmaxf(v, yp[2]);
    v = fmaxf(v, yp[3]);
    v = fmaxf(v, yp[4]);
  } else {
    v = sent[b * 3 + (j - OC_ * POOLED_)];
  }
  z[idx] = fmaxf(v, 0.f);
}

// ---------------- kernel 6: out[b][l] = fc_b[l] + z[b] . fc_w[l] -----------
__global__ __launch_bounds__(256) void k_fc(
    const float* __restrict__ z, const float* __restrict__ fw,
    const float* __restrict__ fb, float* __restrict__ out) {
  int pair = blockIdx.x;        // 12 = B*3, out is [b][l] flat == pair
  int b = pair / 3, l = pair - b * 3;
  const float* zp = z + (size_t)b * FLAT_;
  const float* wp = fw + (size_t)l * FLAT_;
  float s = 0.f;
  for (int i = threadIdx.x; i < FLAT_; i += 256) s += zp[i] * wp[i];
  for (int off = 16; off; off >>= 1) s += __shfl_down(s, off);
  __shared__ float red[8];
  if ((threadIdx.x & 31) == 0) red[threadIdx.x >> 5] = s;
  __syncthreads();
  if (threadIdx.x == 0) {
    float t = 0.f;
    for (int w = 0; w < 8; ++w) t += red[w];
    out[pair] = t + fb[l];
  }
}

extern "C" void kernel_launch(void* const* d_in, const int* in_sizes, int n_in,
                              void* d_out, int out_size, void* d_ws, size_t ws_size,
                              hipStream_t stream) {
  const int*   ids    = (const int*)  d_in[0];
  const float* mask   = (const float*)d_in[1];
  const float* sent   = (const float*)d_in[2];
  const float* logits = (const float*)d_in[3];
  const float* hidden = (const float*)d_in[4];
  const float* convw  = (const float*)d_in[5];
  const float* convb  = (const float*)d_in[6];
  const float* fcw    = (const float*)d_in[7];
  const float* fcb    = (const float*)d_in[8];
  float* out = (float*)d_out;

  char* ws = (char*)d_ws;
  size_t off = 0;
  auto carve = [&](size_t bytes) {
    size_t o = off;
    off = (off + bytes + 255) & ~(size_t)255;
    return o;
  };
  float*          surp = (float*)         (ws + carve((size_t)B_ * S_ * 4));
  unsigned short* xb   = (unsigned short*)(ws + carve((size_t)B_ * S_ * CP_ * 2));
  unsigned short* wA   = (unsigned short*)(ws + carve((size_t)KW_ * OC_ * CP_ * 2));
  float*          y    = (float*)         (ws + carve((size_t)B_ * OC_ * S_ * 4));
  float*          z    = (float*)         (ws + carve((size_t)B_ * FLAT_ * 4));
  (void)in_sizes; (void)n_in; (void)out_size; (void)ws_size;

  k_surprisal<<<B_ * S_, 256, 0, stream>>>(logits, ids, mask, surp);

  {
    int total = B_ * S_ * CP_;
    k_pack_x<<<(total + 255) / 256, 256, 0, stream>>>(hidden, surp, xb);
  }
  {
    int total = KW_ * OC_ * CP_;
    k_pack_w<<<(total + 255) / 256, 256, 0, stream>>>(convw, wA);
  }

  k_conv_wmma<<<dim3(S_ / 32, B_), 256, 0, stream>>>(xb, wA, convb, y);

  {
    int total = B_ * FLAT_;
    k_pool_relu<<<(total + 255) / 256, 256, 0, stream>>>(y, sent, z);
  }

  k_fc<<<12, 256, 0, stream>>>(z, fcw, fcb, out);
}
